// WindowTransExt_79224966742781
// MI455X (gfx1250) — matrix operand order repaired and checked by reading, not compiled, not run on hardware
//
#include <hip/hip_runtime.h>
#include <math.h>

// ---------------------------------------------------------------------------
// Types for CDNA5 WMMA (wave32): v_wmma_f32_16x16x32_f16
// ---------------------------------------------------------------------------
typedef _Float16 h16;
typedef __attribute__((ext_vector_type(16))) _Float16 v16h;
typedef __attribute__((ext_vector_type(8)))  _Float16 v8h;
typedef __attribute__((ext_vector_type(8)))  float    v8f;

#define B_     8
#define H_     112
#define W_     112
#define L_     (H_*W_)        // 12544
#define M_     (B_*L_)        // 100352 (divisible by 128)
#define CEMB   192
#define NHEAD  6
#define HDIM   32
#define NTOK   49             // 7x7 window
#define NWIN   2048           // B * 16 * 16 windows
static constexpr size_t QKV_ONE = (size_t)NWIN * NHEAD * NTOK * HDIM; // elements per q/k/v

enum { EPI_QKV = 0, EPI_PROJ = 1, EPI_FC1 = 2, EPI_FC2 = 3 };

// ---------------------------------------------------------------------------
// Prep kernels.
// Weights are already (N_out, K) row-major == the (n,k) layout the GEMM
// B-tile wants -> straight f32->f16 copy. Conv input goes NCHW -> NHWC f16 and
// conv weight K axis is reordered (ci,kh,kw) -> (tap= kh*3+kw, ci) so that each
// 16-half im2col segment is contiguous in memory (one tap, 16 channels).
// ---------------------------------------------------------------------------
__global__ void f32_to_f16_kernel(const float* __restrict__ in, h16* __restrict__ out, int n) {
  int i = blockIdx.x * 256 + threadIdx.x;
  if (i < n) out[i] = (h16)in[i];
}

__global__ void nchw_to_nhwc_f16_kernel(const float* __restrict__ in, h16* __restrict__ out) {
  int i = blockIdx.x * 256 + threadIdx.x;  // over B*L*64, coalesced writes
  int ci = i & 63;
  int rest = i >> 6;  // b*L + l
  int b = rest / L_, l = rest % L_;
  out[i] = (h16)in[((size_t)b * 64 + ci) * L_ + l];
}

__global__ void conv_w_reorder_kernel(const float* __restrict__ in, h16* __restrict__ out) {
  int i = blockIdx.x * 256 + threadIdx.x;  // over 192*576 (output index)
  if (i < 192 * 576) {
    int n = i / 576, kp = i % 576;
    int tap = kp >> 6, ci = kp & 63;
    out[i] = (h16)in[n * 576 + ci * 9 + tap];
  }
}

// ---------------------------------------------------------------------------
// WMMA fragment helpers (LDS tiles -> register fragments), all-b128 LDS traffic.
// lA[128][40] (m,k); lB[64][40] (n,k); pad 8 halves.
// A frag (16x32): lanes 0-15 row=lane, K {0..7,16..23}; lanes 16-31 K {8..15,24..31}
// B frag (32x16): lanes 0-15 col=lane, K {0..7,16..23}; lanes 16-31 K {8..15,24..31}
// ---------------------------------------------------------------------------
union UV { v16h v; v8h h[2]; };

__device__ __forceinline__ void wmma_step(const h16 (*lA)[40], const h16 (*lB)[40],
                                          int wv, int lane, v8f acc[4]) {
  const int halfsel = lane >> 4, rowin = lane & 15;
  UV ua;
  const int ab = halfsel ? 8 : 0;
  ua.h[0] = *(const v8h*)&lA[wv * 16 + rowin][ab];
  ua.h[1] = *(const v8h*)&lA[wv * 16 + rowin][ab + 16];
  const int kb = halfsel * 8;
#pragma unroll
  for (int s = 0; s < 4; ++s) {
    UV ub;
    ub.h[0] = *(const v8h*)&lB[s * 16 + rowin][kb];
    ub.h[1] = *(const v8h*)&lB[s * 16 + rowin][kb + 16];
    acc[s] = __builtin_amdgcn_wmma_f32_16x16x32_f16(false, ua.v, false, ub.v,
                                                    (short)0, acc[s], false, false);
  }
}

// ---------------------------------------------------------------------------
// Generic f16 WMMA GEMM: C[M,N] = A[M,K] * B^T where Bw is (N,K) row-major.
// Block tile 128x64, K-step 32, 8 waves (wave = 16 rows x 64 cols), LDS
// double-buffered: one barrier per K-step, global prefetch overlaps WMMA.
// ---------------------------------------------------------------------------
template <int EPI>
__global__ __launch_bounds__(256) void gemm_wmma(
    const h16* __restrict__ A, const h16* __restrict__ Bw, int M, int N, int K,
    const float* __restrict__ bias, const float* __restrict__ res0,
    const float* __restrict__ res1, float* __restrict__ outF, h16* __restrict__ outH) {
  __shared__ __align__(16) h16 lA[2][128][40];
  __shared__ __align__(16) h16 lB[2][64][40];
  const int t = threadIdx.x;
  const int wv = t >> 5, lane = t & 31;
  const int halfsel = lane >> 4, rowin = lane & 15;
  const int m0 = blockIdx.x * 128;
  const int n0 = blockIdx.y * 64;
  v8f acc[4] = {};
  const int arow = t >> 1, aseg = t & 1;     // A: 128 rows x (2 x 16-half segs)
  const int brow = t >> 2, bseg = t & 3;     // B: 64 rows x (4 x 8-half segs)
  const h16* aptr = A + (size_t)(m0 + arow) * K + aseg * 16;
  const h16* bptr = Bw + (size_t)(n0 + brow) * K + bseg * 8;

  // preload tile 0
  v8h a0 = *(const v8h*)aptr;
  v8h a1 = *(const v8h*)(aptr + 8);
  v8h b0 = *(const v8h*)bptr;
  *(v8h*)&lA[0][arow][aseg * 16]     = a0;
  *(v8h*)&lA[0][arow][aseg * 16 + 8] = a1;
  *(v8h*)&lB[0][brow][bseg * 8]      = b0;
  __syncthreads();

  const int nk = K >> 5;
  for (int ks = 0; ks < nk; ++ks) {
    const int cur = ks & 1;
    if (ks + 1 < nk) {  // prefetch next tile to registers
      const h16* ap = aptr + (ks + 1) * 32;
      a0 = *(const v8h*)ap;
      a1 = *(const v8h*)(ap + 8);
      b0 = *(const v8h*)(bptr + (ks + 1) * 32);
    }
    wmma_step(lA[cur], lB[cur], wv, lane, acc);
    if (ks + 1 < nk) {
      const int nxt = cur ^ 1;
      *(v8h*)&lA[nxt][arow][aseg * 16]     = a0;
      *(v8h*)&lA[nxt][arow][aseg * 16 + 8] = a1;
      *(v8h*)&lB[nxt][brow][bseg * 8]      = b0;
      __syncthreads();
    }
  }

  // --- fused epilogues ---
#pragma unroll
  for (int s = 0; s < 4; ++s) {
    const int gn = n0 + s * 16 + rowin;
#pragma unroll
    for (int r = 0; r < 8; ++r) {
      const int gm = m0 + wv * 16 + halfsel * 8 + r;
      float val = acc[s][r];
      if (EPI == EPI_QKV) {
        val += bias[gn];
        int which = gn / 192, rem = gn % 192;
        int head = rem >> 5, d = rem & 31;
        if (which == 0) val *= 0.17677669529663687f;  // HD^-0.5
        int win = gm / 49, tok = gm % 49;
        outH[(size_t)which * QKV_ONE + (((size_t)win * 6 + head) * 49 + tok) * 32 + d] =
            (h16)val;
      } else if (EPI == EPI_PROJ) {
        // window-reverse scatter + residual with t0
        int win = gm / 49, tok = gm % 49;
        int b = win >> 8, winl = win & 255;
        int wh = winl >> 4, wwc = winl & 15;
        int r7 = tok / 7, c7 = tok % 7;
        size_t l = (size_t)(wh * 7 + r7) * 112 + (wwc * 7 + c7);
        size_t idx = ((size_t)b * L_ + l) * 192 + gn;
        outF[idx] = val + bias[gn] + res0[idx];
      } else if (EPI == EPI_FC1) {
        val += bias[gn];
        float g = 0.5f * val * (1.0f + erff(val * 0.70710678118654752f));  // exact GELU
        outH[(size_t)gm * N + gn] = (h16)g;
      } else {  // EPI_FC2: + bias + t1 + t0, scatter to NCHW output
        size_t idx = (size_t)gm * 192 + gn;
        float v2 = val + bias[gn] + res0[idx] + res1[idx];
        int b = gm / L_, l = gm % L_;
        outF[((size_t)b * 192 + gn) * L_ + l] = v2;
      }
    }
  }
}

// ---------------------------------------------------------------------------
// Conv 3x3 (pad 1) as NHWC implicit GEMM: M=100352, K=576 (9 taps x 64 ch),
// N=192. With K ordered (tap, ci), each 16-half A segment is one contiguous
// b128-pair load from NHWC x with a single per-tap bounds check.
// ---------------------------------------------------------------------------
__global__ __launch_bounds__(256) void conv_gemm_wmma(
    const h16* __restrict__ xn, const h16* __restrict__ Bw,
    const float* __restrict__ bias, float* __restrict__ t0) {
  __shared__ __align__(16) h16 lA[2][128][40];
  __shared__ __align__(16) h16 lB[2][64][40];
  const int t = threadIdx.x;
  const int wv = t >> 5, lane = t & 31;
  const int halfsel = lane >> 4, rowin = lane & 15;
  const int m0 = blockIdx.x * 128;
  const int n0 = blockIdx.y * 64;
  const int K = 576;
  v8f acc[4] = {};
  const int arow = t >> 1, aseg = t & 1;
  const int brow = t >> 2, bseg = t & 3;
  const int gm = m0 + arow;
  const int bb = gm / L_, l = gm % L_;
  const int hh = l / 112, ww = l % 112;
  const h16* bptr = Bw + (size_t)(n0 + brow) * K + bseg * 8;

  // gather one 16-half im2col segment: tap = kbase/64, channels ci0..ci0+15
  auto gather = [&](int ks, v8h& g0, v8h& g1) {
    const int kbase = ks * 32 + aseg * 16;
    const int tap = kbase >> 6, ci0 = kbase & 63;
    const int kh = tap / 3, kw = tap - kh * 3;
    const int ih = hh + kh - 1, iw = ww + kw - 1;
    if ((unsigned)ih < 112u && (unsigned)iw < 112u) {
      const h16* xp = xn + (((size_t)(bb * 112 + ih)) * 112 + iw) * 64 + ci0;
      g0 = *(const v8h*)xp;
      g1 = *(const v8h*)(xp + 8);
    } else {
      v8h z = {};
      g0 = z;
      g1 = z;
    }
  };

  v8h a0, a1;
  gather(0, a0, a1);
  v8h b0 = *(const v8h*)bptr;
  *(v8h*)&lA[0][arow][aseg * 16]     = a0;
  *(v8h*)&lA[0][arow][aseg * 16 + 8] = a1;
  *(v8h*)&lB[0][brow][bseg * 8]      = b0;
  __syncthreads();

  const int nk = K >> 5;  // 18
  for (int ks = 0; ks < nk; ++ks) {
    const int cur = ks & 1;
    if (ks + 1 < nk) {
      gather(ks + 1, a0, a1);
      b0 = *(const v8h*)(bptr + (ks + 1) * 32);
    }
    wmma_step(lA[cur], lB[cur], wv, lane, acc);
    if (ks + 1 < nk) {
      const int nxt = cur ^ 1;
      *(v8h*)&lA[nxt][arow][aseg * 16]     = a0;
      *(v8h*)&lA[nxt][arow][aseg * 16 + 8] = a1;
      *(v8h*)&lB[nxt][brow][bseg * 8]      = b0;
      __syncthreads();
    }
  }
#pragma unroll
  for (int s = 0; s < 4; ++s) {
    int gn = n0 + s * 16 + rowin;
#pragma unroll
    for (int r = 0; r < 8; ++r) {
      int gmo = m0 + wv * 16 + halfsel * 8 + r;
      t0[(size_t)gmo * 192 + gn] = acc[s][r] + bias[gn];
    }
  }
}

// ---------------------------------------------------------------------------
// LayerNorm over C=192, one wave per token; WIN=true writes windowed order f16.
// ---------------------------------------------------------------------------
template <bool WIN>
__global__ __launch_bounds__(256) void ln_kernel(const float* __restrict__ tin,
                                                 const float* __restrict__ w,
                                                 const float* __restrict__ bp,
                                                 h16* __restrict__ dst) {
  const int wv = threadIdx.x >> 5, lane = threadIdx.x & 31;
  const int m = blockIdx.x * 8 + wv;
  const float* row = tin + (size_t)m * 192;
  float v[6];
  float s = 0.f, s2 = 0.f;
#pragma unroll
  for (int i = 0; i < 6; ++i) {
    v[i] = row[lane + 32 * i];
    s += v[i];
    s2 += v[i] * v[i];
  }
#pragma unroll
  for (int o = 16; o > 0; o >>= 1) {
    s += __shfl_xor(s, o, 32);
    s2 += __shfl_xor(s2, o, 32);
  }
  float mean = s * (1.f / 192.f);
  float var = s2 * (1.f / 192.f) - mean * mean;
  float inv = rsqrtf(var + 1e-5f);
  size_t drow;
  if (WIN) {
    int b = m / L_, l = m % L_;
    int hh = l / 112, ww = l % 112;
    int wh = hh / 7, r7 = hh % 7;
    int wc = ww / 7, c7 = ww % 7;
    int win = (b * 16 + wh) * 16 + wc;
    drow = (size_t)win * 49 + (r7 * 7 + c7);
  } else {
    drow = (size_t)m;
  }
#pragma unroll
  for (int i = 0; i < 6; ++i) {
    int c = lane + 32 * i;
    dst[drow * 192 + c] = (h16)((v[i] - mean) * inv * w[c] + bp[c]);
  }
}

// ---------------------------------------------------------------------------
// Windowed attention: one block per window, wave-per-head; q/k/v staged in LDS
// with b128 copies. 49x49 scores + rel-pos bias + softmax + AV in f32.
// ---------------------------------------------------------------------------
__global__ __launch_bounds__(256) void attn_kernel(const h16* __restrict__ qkv,
                                                   const float* __restrict__ rel_table,
                                                   const int* __restrict__ rel_index,
                                                   h16* __restrict__ o16) {
  __shared__ __align__(16) h16 sQ[NHEAD * NTOK * HDIM];
  __shared__ __align__(16) h16 sK[NHEAD * NTOK * HDIM];
  __shared__ __align__(16) h16 sV[NHEAD * NTOK * HDIM];
  __shared__ float sP[8][64];
  const int win = blockIdx.x;
  const int nElems = NHEAD * NTOK * HDIM;  // 9408
  const int nVec = nElems / 8;             // 1176 v8h
  const v8h* qg = (const v8h*)(qkv + (size_t)win * nElems);
  const v8h* kg = (const v8h*)(qkv + QKV_ONE + (size_t)win * nElems);
  const v8h* vg = (const v8h*)(qkv + 2 * QKV_ONE + (size_t)win * nElems);
  v8h* sQ8 = (v8h*)sQ;
  v8h* sK8 = (v8h*)sK;
  v8h* sV8 = (v8h*)sV;
  for (int i = threadIdx.x; i < nVec; i += 256) {
    sQ8[i] = qg[i];
    sK8[i] = kg[i];
    sV8[i] = vg[i];
  }
  __syncthreads();
  const int wv = threadIdx.x >> 5, lane = threadIdx.x & 31;
  for (int h = wv; h < NHEAD; h += 8) {
    const h16* Q = sQ + h * NTOK * HDIM;
    const h16* Kp = sK + h * NTOK * HDIM;
    const h16* Vp = sV + h * NTOK * HDIM;
    const int j1 = lane + 32;
    const bool j1v = (j1 < NTOK);
    for (int i = 0; i < NTOK; ++i) {
      float s0 = 0.f, s1 = 0.f;
      for (int d = 0; d < HDIM; ++d) {
        float qd = (float)Q[i * HDIM + d];
        s0 = fmaf(qd, (float)Kp[lane * HDIM + d], s0);
        s1 = fmaf(qd, (float)Kp[(j1v ? j1 : 0) * HDIM + d], s1);
      }
      s0 += rel_table[rel_index[i * NTOK + lane] * NHEAD + h];
      s1 = j1v ? (s1 + rel_table[rel_index[i * NTOK + j1] * NHEAD + h]) : -3.0e38f;
      float mx = fmaxf(s0, s1);
#pragma unroll
      for (int o = 16; o > 0; o >>= 1) mx = fmaxf(mx, __shfl_xor(mx, o, 32));
      float e0 = __expf(s0 - mx);
      float e1 = j1v ? __expf(s1 - mx) : 0.f;
      float sum = e0 + e1;
#pragma unroll
      for (int o = 16; o > 0; o >>= 1) sum += __shfl_xor(sum, o, 32);
      float inv = 1.f / sum;
      sP[wv][lane] = e0 * inv;
      if (j1v) sP[wv][j1] = e1 * inv;
      float accv = 0.f;
      for (int j = 0; j < NTOK; ++j) accv = fmaf(sP[wv][j], (float)Vp[j * HDIM + lane], accv);
      o16[((size_t)win * 49 + i) * 192 + h * 32 + lane] = (h16)accv;
    }
  }
}

// ---------------------------------------------------------------------------
// Host-side orchestration
// ---------------------------------------------------------------------------
extern "C" void kernel_launch(void* const* d_in, const int* in_sizes, int n_in,
                              void* d_out, int out_size, void* d_ws, size_t ws_size,
                              hipStream_t stream) {
  (void)in_sizes; (void)n_in; (void)out_size; (void)ws_size;
  const float* x         = (const float*)d_in[0];
  const float* proj_w    = (const float*)d_in[1];
  const float* proj_b    = (const float*)d_in[2];
  const float* ln1_w     = (const float*)d_in[3];
  const float* ln1_b     = (const float*)d_in[4];
  const float* qkv_w     = (const float*)d_in[5];
  const float* qkv_b     = (const float*)d_in[6];
  const float* rel_table = (const float*)d_in[7];
  const float* attn_pw   = (const float*)d_in[8];
  const float* attn_pb   = (const float*)d_in[9];
  const float* ln2_w     = (const float*)d_in[10];
  const float* ln2_b     = (const float*)d_in[11];
  const float* fc1_w     = (const float*)d_in[12];
  const float* fc1_b     = (const float*)d_in[13];
  const float* fc2_w     = (const float*)d_in[14];
  const float* fc2_b     = (const float*)d_in[15];
  const int*   rel_idx   = (const int*)d_in[16];

  char* ws = (char*)d_ws;
  size_t off = 0;
  auto take = [&](size_t bytes) -> char* {
    char* p = ws + off;
    off += (bytes + 255) & ~(size_t)255;
    return p;
  };
  h16* xn     = (h16*)take((size_t)B_ * 64 * H_ * W_ * 2);  // NHWC f16
  h16* wCONV  = (h16*)take((size_t)192 * 576 * 2);   // (N=192, K=tap*64+ci)
  h16* wQKV   = (h16*)take((size_t)576 * 192 * 2);   // (N=576, K=192) — native layout
  h16* wPROJ  = (h16*)take((size_t)192 * 192 * 2);
  h16* wFC1   = (h16*)take((size_t)768 * 192 * 2);
  h16* wFC2   = (h16*)take((size_t)192 * 768 * 2);
  float* t0   = (float*)take((size_t)M_ * 192 * 4);  // conv out, (B,L,C)
  float* t1   = (float*)take((size_t)M_ * 192 * 4);  // after attn residual
  h16* a16    = (h16*)take((size_t)M_ * 192 * 2);    // LN outputs (reused)
  h16* o16    = (h16*)take((size_t)M_ * 192 * 2);    // attention output
  h16* big    = (h16*)take((size_t)M_ * 768 * 2);    // qkv (115.6MB) then fc1 out (154MB)
  h16* qkv16  = big;
  h16* m16    = big;
  float* outp = (float*)d_out;

  const int nx = B_ * 64 * H_ * W_;
  nchw_to_nhwc_f16_kernel<<<(nx + 255) / 256, 256, 0, stream>>>(x, xn);
  conv_w_reorder_kernel<<<(192 * 576 + 255) / 256, 256, 0, stream>>>(proj_w, wCONV);
  f32_to_f16_kernel<<<(576 * 192 + 255) / 256, 256, 0, stream>>>(qkv_w, wQKV, 576 * 192);
  f32_to_f16_kernel<<<(192 * 192 + 255) / 256, 256, 0, stream>>>(attn_pw, wPROJ, 192 * 192);
  f32_to_f16_kernel<<<(768 * 192 + 255) / 256, 256, 0, stream>>>(fc1_w, wFC1, 768 * 192);
  f32_to_f16_kernel<<<(192 * 768 + 255) / 256, 256, 0, stream>>>(fc2_w, wFC2, 192 * 768);

  const int MB = M_ / 128;  // 784
  conv_gemm_wmma<<<dim3(MB, 3), 256, 0, stream>>>(xn, wCONV, proj_b, t0);
  ln_kernel<true><<<M_ / 8, 256, 0, stream>>>(t0, ln1_w, ln1_b, a16);
  gemm_wmma<EPI_QKV><<<dim3(MB, 9), 256, 0, stream>>>(a16, wQKV, M_, 576, 192, qkv_b,
                                                      nullptr, nullptr, nullptr, qkv16);
  attn_kernel<<<NWIN, 256, 0, stream>>>(qkv16, rel_table, rel_idx, o16);
  gemm_wmma<EPI_PROJ><<<dim3(MB, 3), 256, 0, stream>>>(o16, wPROJ, M_, 192, 192, attn_pb,
                                                       t0, nullptr, t1, nullptr);
  ln_kernel<false><<<M_ / 8, 256, 0, stream>>>(t1, ln2_w, ln2_b, a16);
  gemm_wmma<EPI_FC1><<<dim3(MB, 12), 256, 0, stream>>>(a16, wFC1, M_, 768, 192, fc1_b,
                                                       nullptr, nullptr, nullptr, m16);
  gemm_wmma<EPI_FC2><<<dim3(MB, 3), 256, 0, stream>>>(m16, wFC2, M_, 192, 768, fc2_b,
                                                      t0, t1, outp, nullptr);
}
